// SSBlock_52063593562539
// MI455X (gfx1250) — matrix operand order repaired and checked by reading, not compiled, not run on hardware
//
#include <hip/hip_runtime.h>

typedef __attribute__((ext_vector_type(16))) _Float16 v16h;
typedef __attribute__((ext_vector_type(8)))  float    v8f;

#define NPIX   16384          // 128*128
#define NTOK   4096           // 64*64 tokens per window
#define CCH    64

#if __has_builtin(__builtin_amdgcn_exp2f)
#define EXP2(x) __builtin_amdgcn_exp2f(x)     // bare v_exp_f32, no range fixup
#else
#define EXP2(x) exp2f(x)
#endif

// ---------------- Stage 1: deterministic LayerNorm statistics ----------------
__global__ void ss_reduce1(const float* __restrict__ x, float* __restrict__ partial) {
    __shared__ float s1[256], s2[256];
    int tid = threadIdx.x;
    float a = 0.f, b = 0.f;
    for (int i = blockIdx.x * 256 + tid; i < (1 << 20); i += 256 * 1024) {
        float v = x[i]; a += v; b += v * v;
    }
    s1[tid] = a; s2[tid] = b; __syncthreads();
    for (int st = 128; st > 0; st >>= 1) {
        if (tid < st) { s1[tid] += s1[tid + st]; s2[tid] += s2[tid + st]; }
        __syncthreads();
    }
    if (tid == 0) { partial[blockIdx.x] = s1[0]; partial[1024 + blockIdx.x] = s2[0]; }
}

__global__ void ss_reduce2(const float* __restrict__ partial, float* __restrict__ stats) {
    __shared__ float s1[1024], s2[1024];
    int tid = threadIdx.x;
    s1[tid] = partial[tid]; s2[tid] = partial[1024 + tid]; __syncthreads();
    for (int st = 512; st > 0; st >>= 1) {
        if (tid < st) { s1[tid] += s1[tid + st]; s2[tid] += s2[tid + st]; }
        __syncthreads();
    }
    if (tid == 0) {
        float mean = s1[0] * (1.0f / 1048576.0f);
        float var  = s2[0] * (1.0f / 1048576.0f) - mean * mean;
        stats[0] = mean;
        stats[1] = rsqrtf(var + 1e-5f);
    }
}

// ------ Stage 2: qk = normalize(LN(x)_win @ wqk) [token-major],
//                 v  = x_win stored TRANSPOSED (embed-major) for PV fragments --
__global__ void ss_buildqk(const float* __restrict__ x, const float* __restrict__ wqk,
                           const float* __restrict__ stats,
                           _Float16* __restrict__ qkbuf, _Float16* __restrict__ vbufT) {
    __shared__ float xin[64];
    __shared__ float red[64];
    int row = blockIdx.x;                 // b2*4096 + n
    int b2 = row >> 12, n = row & 4095;
    int dh = b2 >> 1, dw = b2 & 1;
    int hp = n >> 6,  wp = n & 63;
    int h = hp * 2 + dh, w = wp * 2 + dw;
    int e = threadIdx.x;

    float mean = stats[0], rstd = stats[1];
    float xv = x[(e * 128 + h) * 128 + w];          // channel c == thread id
    xin[e] = (xv - mean) * rstd;
    vbufT[((size_t)b2 * 64 + e) * NTOK + n] = (_Float16)xv;   // V^T, raw x
    __syncthreads();

    float acc = 0.f;
    #pragma unroll
    for (int c = 0; c < 64; ++c) acc = fmaf(xin[c], wqk[c * 64 + e], acc);

    red[e] = acc * acc; __syncthreads();
    for (int st = 32; st > 0; st >>= 1) {
        if (e < st) red[e] += red[e + st];
        __syncthreads();
    }
    float norm = sqrtf(red[0]) + 1e-8f;
    qkbuf[(size_t)row * 64 + e] = (_Float16)(acc / norm);
}

// ---- Stage 3: transposed attention, one wave / 16-row tile, WMMA f16 ---------
// S^T = K·Q^T  (keys are the M dim); qk rows are unit vectors so the score
// t = ((dot+1)/8)*log2e is in [0, 0.362]: exact softmax needs NO max shift,
// no rescaling, no online machinery. l accumulates per lane; one shfl at end.
// O^T = V^T·P^T (V stored embed-major; P staged through LDS row-major).
__global__ __launch_bounds__(32)
void ss_attn(const _Float16* __restrict__ qk, const _Float16* __restrict__ vT,
             float* __restrict__ outf) {
    __shared__ _Float16 sP[16 * 32];              // P tile: [row][key] f16
    const int b2 = blockIdx.x >> 8;               // 256 row-tiles per window
    const int r0 = (blockIdx.x & 255) * 16;
    const int lane = threadIdx.x;
    const int r = lane & 15, hh = lane >> 4;
    const _Float16* qb = qk + (size_t)b2 * NTOK * 64;
    const _Float16* vb = vT + (size_t)b2 * 64 * NTOK;

    // B operand = Q^T (e x rows): loop-invariant, loaded once.
    // B layout: N = lane%16 (row), K = (v>>2)*16 + hh*8 + 2(v&3) (+1)  (e index)
    v16h qB[2];
    #pragma unroll
    for (int f = 0; f < 2; ++f)
        #pragma unroll
        for (int vv = 0; vv < 8; ++vv) {
            int e0 = f * 32 + (vv >> 2) * 16 + hh * 8 + (vv & 3) * 2;
            const _Float16* p = qb + (size_t)(r0 + r) * 64 + e0;
            qB[f][2 * vv]     = p[0];
            qB[f][2 * vv + 1] = p[1];
        }

    v8f oacc[4] = {};                    // O^T: [e-tile][comp] , row = lane%16
    float l = 0.f;                       // per-lane partial softmax denominator
    const float C1 = 0.125f * 1.44269504088896340736f;   // (1/8)*log2(e)

    for (int j = 0; j < NTOK; j += 32) {
        // ---- S^T = K·Q^T : two 16-key tiles, K-dim(e) = 64 = 2x32
        v8f s[2];
        #pragma unroll
        for (int kt = 0; kt < 2; ++kt) {
            v8f acc = {};
            #pragma unroll
            for (int f = 0; f < 2; ++f) {
                v16h ka;                 // A: 16 keys x 32 e
                #pragma unroll
                for (int vv = 0; vv < 8; ++vv) {
                    int e0 = f * 32 + (vv >> 2) * 16 + hh * 8 + (vv & 3) * 2;
                    const _Float16* p = qb + (size_t)(j + kt * 16 + r) * 64 + e0;
                    ka[2 * vv]     = p[0];
                    ka[2 * vv + 1] = p[1];
                }
                acc = __builtin_amdgcn_wmma_f32_16x16x32_f16(
                        false, ka, false, qB[f], (short)0, acc, false, false);
            }
            s[kt] = acc;
        }

        // ---- exact softmax numerator: p = exp2((dot+1)/8 * log2e), bounded [1,1.29]
        float p0[8], p1[8];
        #pragma unroll
        for (int c = 0; c < 8; ++c) {
            p0[c] = EXP2(fmaf(s[0][c], C1, C1));
            p1[c] = EXP2(fmaf(s[1][c], C1, C1));
            l += p0[c] + p1[c];
        }

        // ---- stage P [row][key] into LDS; contiguous 8-half runs per lane
        #pragma unroll
        for (int c = 0; c < 8; ++c) {
            sP[r * 32 + 8 * hh + c]      = (_Float16)p0[c];   // keys kt=0
            sP[r * 32 + 16 + 8 * hh + c] = (_Float16)p1[c];   // keys kt=1
        }
        asm volatile("s_wait_dscnt 0" ::: "memory");   // single-wave LDS RAW fence

        // B operand = P^T (32 keys x 16 rows): B[k][r] = sP[r*32+k]
        v16h pb;
        #pragma unroll
        for (int vv = 0; vv < 8; ++vv) {
            int k0 = (vv >> 2) * 16 + hh * 8 + (vv & 3) * 2;
            pb[2 * vv]     = sP[r * 32 + k0];
            pb[2 * vv + 1] = sP[r * 32 + k0 + 1];
        }

        // ---- O^T += V^T·P^T : A = V^T tile (16 e x 32 keys), contiguous keys
        #pragma unroll
        for (int et = 0; et < 4; ++et) {
            v16h av;
            #pragma unroll
            for (int vv = 0; vv < 8; ++vv) {
                int koff = (vv >> 2) * 16 + hh * 8 + (vv & 3) * 2;
                const _Float16* p = vb + (size_t)(et * 16 + r) * NTOK + j + koff;
                av[2 * vv]     = p[0];
                av[2 * vv + 1] = p[1];
            }
            oacc[et] = __builtin_amdgcn_wmma_f32_16x16x32_f16(
                        false, av, false, pb, (short)0, oacc[et], false, false);
        }
    }

    // ---- epilogue: combine half-wave partial sums, O /= l, store
    l += __shfl_xor(l, 16, 32);          // lanes r and r+16 partition the keys
    float inv = 1.0f / l;
    float* ob = outf + (size_t)b2 * NTOK * 64;
    #pragma unroll
    for (int et = 0; et < 4; ++et)
        #pragma unroll
        for (int c = 0; c < 8; ++c)
            ob[(size_t)(r0 + r) * 64 + et * 16 + 8 * hh + c] = oacc[et][c] * inv;
}

// -------- Stage 4: convl (1x1, 65->64 with unet as ch 64) + residual with x ---
__global__ void ss_convl_add(const float* __restrict__ x, const float* __restrict__ attn,
                             const float* __restrict__ unet, const float* __restrict__ wl,
                             const float* __restrict__ bl, float* __restrict__ t0) {
    int id = blockIdx.x * 256 + threadIdx.x;      // o*16384 + p
    int o = id >> 14, p = id & (NPIX - 1);
    const float* w = wl + o * 65;
    float acc = bl[o] + w[64] * unet[p];
    #pragma unroll 8
    for (int i = 0; i < 64; ++i) acc = fmaf(w[i], attn[i * NPIX + p], acc);
    t0[id] = x[id] + acc;
}

// -------- Stage 5/6: 3x3 conv, LDS-tiled 16x16 pixels, templated dilation -----
template <int DIL, bool RELU>
__global__ void ss_conv3x3(const float* __restrict__ in, const float* __restrict__ w,
                           const float* __restrict__ b, float* __restrict__ out) {
    const int EXT = 16 + 2 * DIL;
    __shared__ float tile[(16 + 2 * DIL) * (16 + 2 * DIL)];
    int o = blockIdx.x >> 6;                      // 64 tiles (8x8) per out channel
    int t = blockIdx.x & 63;
    int th = (t >> 3) * 16, tw = (t & 7) * 16;
    int ly = threadIdx.x >> 4, lx = threadIdx.x & 15;
    float acc = b[o];
    for (int i = 0; i < CCH; ++i) {
        for (int idx = threadIdx.x; idx < EXT * EXT; idx += 256) {
            int yy = th + idx / EXT - DIL;
            int xx = tw + idx % EXT - DIL;
            float vv = 0.f;
            if (yy >= 0 && yy < 128 && xx >= 0 && xx < 128)
                vv = in[(i * 128 + yy) * 128 + xx];
            tile[idx] = vv;
        }
        __syncthreads();
        const float* wp = w + (o * CCH + i) * 9;
        #pragma unroll
        for (int kh = 0; kh < 3; ++kh)
            #pragma unroll
            for (int kw = 0; kw < 3; ++kw)
                acc = fmaf(wp[kh * 3 + kw], tile[(ly + kh * DIL) * EXT + (lx + kw * DIL)], acc);
        __syncthreads();
    }
    if (RELU) acc = fmaxf(acc, 0.f);
    out[(o * 128 + th + ly) * 128 + tw + lx] = acc;
}

// -------- Stage 7: conv3 (1x1) + outer residual --------------------------------
__global__ void ss_conv3_res(const float* __restrict__ x, const float* __restrict__ t2,
                             const float* __restrict__ w3, const float* __restrict__ b3,
                             float* __restrict__ out) {
    int id = blockIdx.x * 256 + threadIdx.x;
    int o = id >> 14, p = id & (NPIX - 1);
    float acc = b3[o];
    #pragma unroll 8
    for (int i = 0; i < 64; ++i) acc = fmaf(w3[o * 64 + i], t2[i * NPIX + p], acc);
    out[id] = x[id] + acc;
}

extern "C" void kernel_launch(void* const* d_in, const int* in_sizes, int n_in,
                              void* d_out, int out_size, void* d_ws, size_t ws_size,
                              hipStream_t stream) {
    const float* x     = (const float*)d_in[0];
    const float* unet  = (const float*)d_in[1];
    const float* wqk   = (const float*)d_in[2];
    const float* w1    = (const float*)d_in[3];
    const float* b1    = (const float*)d_in[4];
    const float* w2    = (const float*)d_in[5];
    const float* b2v   = (const float*)d_in[6];
    const float* w3    = (const float*)d_in[7];
    const float* b3    = (const float*)d_in[8];
    const float* wl    = (const float*)d_in[9];
    const float* bl    = (const float*)d_in[10];
    float* out = (float*)d_out;

    char* ws = (char*)d_ws;
    float*     partial = (float*)(ws + 0);                 // 2048 f32
    float*     stats   = (float*)(ws + 8192);              // mean, rstd
    _Float16*  qkbuf   = (_Float16*)(ws + 8448);           // 4*4096*64 f16 (2 MiB)
    _Float16*  vbufT   = (_Float16*)(ws + 8448 + 2097152); // V^T, 2 MiB
    float*     attnf   = (float*)(ws + 8448 + 2 * 2097152);          // 4 MiB
    float*     t0      = (float*)(ws + 8448 + 2 * 2097152 + 4194304);
    float*     t1      = (float*)(ws + 8448 + 2 * 2097152 + 2 * 4194304);
    float*     t2      = (float*)(ws + 8448 + 2 * 2097152 + 3 * 4194304);

    ss_reduce1<<<1024, 256, 0, stream>>>(x, partial);
    ss_reduce2<<<1, 1024, 0, stream>>>(partial, stats);
    ss_buildqk<<<4 * NTOK, 64, 0, stream>>>(x, wqk, stats, qkbuf, vbufT);
    ss_attn<<<4 * 256, 32, 0, stream>>>(qkbuf, vbufT, attnf);
    ss_convl_add<<<4096, 256, 0, stream>>>(x, attnf, unet, wl, bl, t0);
    ss_conv3x3<1, true><<<4096, 256, 0, stream>>>(t0, w1, b1, t1);
    ss_conv3x3<2, true><<<4096, 256, 0, stream>>>(t1, w2, b2v, t2);
    ss_conv3_res<<<4096, 256, 0, stream>>>(x, t2, w3, b3, out);
}
